// Block_15006615734251
// MI455X (gfx1250) — compile-verified, matrix operand order checked by reading, not run on hardware
//
#include <hip/hip_runtime.h>
#include <hip/hip_bf16.h>
#include <math.h>

#define TOK  8192
#define NSEQ 4096
#define CDIM 384
#define NH   6
#define HDIM 64
#define HALF 192
#define MLPH 1536

typedef __attribute__((ext_vector_type(16))) __bf16 v16bf;
typedef __attribute__((ext_vector_type(8)))  float  v8f;
typedef __attribute__((ext_vector_type(4)))  unsigned int v4u;
typedef __attribute__((ext_vector_type(4)))  int v4i;
typedef __attribute__((ext_vector_type(8)))  int v8i;

#if __has_builtin(__builtin_amdgcn_tensor_load_to_lds) && __has_builtin(__builtin_amdgcn_s_wait_tensorcnt)
#define HAVE_TDM 1
#else
#define HAVE_TDM 0
#warning "gfx1250 TDM builtins unavailable; using cooperative LDS copy fallback"
#endif

__device__ __forceinline__ unsigned short bf16_bits(float f) {
  union { float f; unsigned int u; } c; c.f = f;
  unsigned int u = c.u;
  return (unsigned short)((u + 0x7FFFu + ((u >> 16) & 1u)) >> 16);
}

// A-fragment (and, by layout symmetry, B-fragment of the transpose) of a
// row-major bf16 matrix [rows][ld] at (row0, k0): 16x32 tile, K=32.
__device__ __forceinline__ v16bf load_frag(const unsigned short* __restrict__ base,
                                           int ld, int row0, int k0) {
  int lane = threadIdx.x & 31;
  const unsigned short* p = base + (size_t)(row0 + (lane & 15)) * ld + k0 + ((lane >> 4) << 3);
  union { v16bf v; uint4 q[2]; } f;
  f.q[0] = *(const uint4*)p;        // K = ko .. ko+7   -> elems 0-7
  f.q[1] = *(const uint4*)(p + 16); // K = ko+16..ko+23 -> elems 8-15
  return f.v;
}

__device__ __forceinline__ v8f wmma_bf16(v16bf a, v16bf b, v8f c) {
  return __builtin_amdgcn_wmma_f32_16x16x32_bf16(false, a, false, b, (short)0, c, false, false);
}

#if HAVE_TDM
// Tensor DMA: 2-D tile of bf16 (data_size=2B) from row-major tensor
// [tdim1][tdim0] (element stride0 between rows) -> compacted LDS.
// D# packing per CDNA5 ISA.
__device__ __forceinline__ void tdm_load_2d(void* lds, const void* gptr,
                                            unsigned tdim0, unsigned tdim1,
                                            unsigned stride0,
                                            unsigned tile0, unsigned tile1) {
  unsigned lds_addr = (unsigned)(unsigned long long)lds;         // LDS aperture: low 32b = LDS byte addr
  unsigned long long ga = (unsigned long long)gptr;
  v4u g0;
  g0[0] = 1u;                                                    // count=1, is_restore=0, gather=0
  g0[1] = lds_addr;                                              // lds_addr [63:32]
  g0[2] = (unsigned)ga;                                          // global_addr [95:64]
  g0[3] = ((unsigned)(ga >> 32) & 0x01FFFFFFu) | 0x80000000u;    // addr[56:32] | type=2<<30
  v8i g1;
  g1[0] = (int)(1u << 16);                                       // wg_mask=0, data_size=1 (2B)
  g1[1] = (int)((tdim0 & 0xFFFFu) << 16);                        // tensor_dim0[15:0] @ bit48
  g1[2] = (int)((tdim0 >> 16) | ((tdim1 & 0xFFFFu) << 16));      // tdim0[31:16] | tdim1[15:0]
  g1[3] = (int)((tdim1 >> 16) | (tile0 << 16));                  // tdim1[31:16] | tile_dim0
  g1[4] = (int)tile1;                                            // tile_dim1 | tile_dim2(=0)<<16
  g1[5] = (int)stride0;                                          // tensor_dim0_stride[31:0]
  g1[6] = 0;                                                     // stride0[47:32] | stride1[15:0]
  g1[7] = 0;                                                     // stride1[47:16]
  v4i gz = {0, 0, 0, 0};
#if __clang_major__ >= 23
  v8i gz8 = {0, 0, 0, 0, 0, 0, 0, 0};
  __builtin_amdgcn_tensor_load_to_lds(g0, g1, gz, gz, gz8, 0);
#else
  __builtin_amdgcn_tensor_load_to_lds(g0, g1, gz, gz, 0);
#endif
}
#endif

// Stage one attention chunk: K rows [m0,m0+32) x 64 dims (contiguous 4KB) and
// V^T tile (64 rows x 32 cols out of [64][NSEQ]) into LDS.
__device__ __forceinline__ void stage_tiles(unsigned short* kt, unsigned short* vtl,
                                            const unsigned short* __restrict__ K,
                                            const unsigned short* __restrict__ V,
                                            int m0, int warp, int tid) {
#if HAVE_TDM
  if (warp == 0) {
    tdm_load_2d(kt, K + (size_t)m0 * HDIM, NSEQ * HDIM, 1, NSEQ * HDIM, 2048u, 1u);
    tdm_load_2d(vtl, V + m0, NSEQ, HDIM, NSEQ, 32u, 64u);
  }
#else
  const uint4* gk = (const uint4*)(K + (size_t)m0 * HDIM);
  uint4* lk = (uint4*)kt;
  lk[tid] = gk[tid];
  lk[tid + 128] = gk[tid + 128];
  uint4* lv = (uint4*)vtl;
#pragma unroll
  for (int j = 0; j < 2; ++j) {
    int c = tid + 128 * j;
    lv[c] = *(const uint4*)(V + (size_t)(c >> 2) * NSEQ + m0 + (c & 3) * 8);
  }
#endif
}

// -------------------- elementwise fp32 -> bf16 convert --------------------
__global__ void k_convert(const float* __restrict__ s, unsigned short* __restrict__ d, int n) {
  int i = blockIdx.x * blockDim.x + threadIdx.x;
  int st = gridDim.x * blockDim.x;
  for (; i < n; i += st) d[i] = bf16_bits(s[i]);
}

// -------------------- LayerNorm (one wave per token) --------------------
__global__ void k_ln(const float* __restrict__ x, const float* __restrict__ g,
                     const float* __restrict__ bta, unsigned short* __restrict__ o) {
  int warp = threadIdx.x >> 5, lane = threadIdx.x & 31;
  int t = blockIdx.x * 8 + warp;
  const float* xr = x + (size_t)t * CDIM;
  float v[12]; float s = 0.f;
#pragma unroll
  for (int i = 0; i < 12; ++i) { v[i] = xr[i * 32 + lane]; s += v[i]; }
#pragma unroll
  for (int off = 16; off; off >>= 1) s += __shfl_xor(s, off, 32);
  float m = s * (1.0f / CDIM);
  float q = 0.f;
#pragma unroll
  for (int i = 0; i < 12; ++i) { float d = v[i] - m; q += d * d; }
#pragma unroll
  for (int off = 16; off; off >>= 1) q += __shfl_xor(q, off, 32);
  float rs = rsqrtf(q * (1.0f / CDIM) + 1e-5f);
  unsigned short* orow = o + (size_t)t * CDIM;
#pragma unroll
  for (int i = 0; i < 12; ++i) {
    int c = i * 32 + lane;
    orow[c] = bf16_bits((v[i] - m) * rs * g[c] + bta[c]);
  }
}

// -------------------- generic 16x64 WMMA tile:  acc += A[16xK] * W[64xK]^T ----
__device__ __forceinline__ void gemm16x64(const unsigned short* __restrict__ A, int lda, int row0,
                                          const unsigned short* __restrict__ W, int ldw, int col0,
                                          int ksteps, v8f acc[4]) {
  for (int ks = 0; ks < ksteps; ++ks) {
    int k0 = ks * 32;
    v16bf a = load_frag(A, lda, row0, k0);
#pragma unroll
    for (int f = 0; f < 4; ++f) {
      v16bf w = load_frag(W, ldw, col0 + f * 16, k0);
      acc[f] = wmma_bf16(a, w, acc[f]);
    }
  }
}

// -------------------- QKV GEMM + scatter into head-major q/k and pose v^T ----
__global__ void k_qkv(const unsigned short* __restrict__ hb, const unsigned short* __restrict__ wq,
                      unsigned short* __restrict__ qb, unsigned short* __restrict__ kb,
                      unsigned short* __restrict__ vt) {
  int warp = threadIdx.x >> 5, lane = threadIdx.x & 31;
  int wid = blockIdx.x * 8 + warp;
  int rowt = wid / 18, cgrp = wid % 18;
  int row0 = rowt * 16, col0 = cgrp * 64;
  v8f z = {}; v8f acc[4] = {z, z, z, z};
  gemm16x64(hb, CDIM, row0, wq, CDIM, col0, 12, acc);
  int h16 = (lane >> 4) << 3;
#pragma unroll
  for (int f = 0; f < 4; ++f) {
    int o = col0 + f * 16 + (lane & 15);
    int qkvi = o / CDIM;
    int hh = (o / HDIM) % NH;
    int d = o & (HDIM - 1);
#pragma unroll
    for (int e = 0; e < 8; ++e) {
      int t = row0 + e + h16;
      int b = t >> 12, n = t & (NSEQ - 1);
      unsigned short val = bf16_bits(acc[f][e]);
      if (qkvi == 0)       qb[((size_t)(b * NH + hh) * NSEQ + n) * HDIM + d] = val;
      else if (qkvi == 1)  kb[((size_t)(b * NH + hh) * NSEQ + n) * HDIM + d] = val;
      else if (hh >= 3)    vt[((size_t)(b * 3 + hh - 3) * HDIM + d) * NSEQ + n] = val;
    }
  }
}

// ---- Pass A: attn_obj = sigmoid(Qo Ko^T/8) -> d_out;  ctx1^T = (attn_obj @ Vp)^T
__global__ void k_attn_obj(const unsigned short* __restrict__ qb, const unsigned short* __restrict__ kb,
                           const unsigned short* __restrict__ vt, float* __restrict__ attn,
                           unsigned short* __restrict__ c1t) {
  __shared__ __align__(16) unsigned short ktile[2][32 * 64];
  __shared__ __align__(16) unsigned short vtile[2][64 * 32];
  __shared__ __align__(16) unsigned short pt[4 * 16 * 40];
  int tid = threadIdx.x;
  int warp = tid >> 5, lane = tid & 31;
  int rowt = blockIdx.x & 63;
  int hb_ = blockIdx.x >> 6;              // b*3+ph
  int ph = hb_ % 3, b = hb_ / 3;
  int row0 = rowt * 64 + warp * 16;
  const unsigned short* Q = qb + (size_t)(b * NH + ph) * NSEQ * HDIM;
  const unsigned short* K = kb + (size_t)(b * NH + ph) * NSEQ * HDIM;
  const unsigned short* V = vt + (size_t)(b * 3 + ph) * HDIM * NSEQ;
  v16bf qa0 = load_frag(Q, HDIM, row0, 0);
  v16bf qa1 = load_frag(Q, HDIM, row0, 32);
  v8f z = {}; v8f acc[4] = {z, z, z, z};
  int h16 = (lane >> 4) << 3;
  unsigned short* pw = &pt[warp * 640];
  int cur = 0;
  stage_tiles(&ktile[0][0], &vtile[0][0], K, V, 0, warp, tid);
  for (int m0 = 0; m0 < NSEQ; m0 += 32) {
    int nxt = cur ^ 1;
    if (m0 + 32 < NSEQ) {
      stage_tiles(&ktile[nxt][0], &vtile[nxt][0], K, V, m0 + 32, warp, tid);
#if HAVE_TDM
      __builtin_amdgcn_s_wait_tensorcnt((short)2);
#endif
    } else {
#if HAVE_TDM
      __builtin_amdgcn_s_wait_tensorcnt((short)0);
#endif
    }
    __syncthreads();
    const unsigned short* kt = &ktile[cur][0];
    const unsigned short* vl = &vtile[cur][0];
#pragma unroll
    for (int cg = 0; cg < 2; ++cg) {
      v8f s = z;
      s = wmma_bf16(qa0, load_frag(kt, HDIM, cg * 16, 0), s);
      s = wmma_bf16(qa1, load_frag(kt, HDIM, cg * 16, 32), s);
      int col = m0 + cg * 16 + (lane & 15);
#pragma unroll
      for (int e = 0; e < 8; ++e) {
        float p = 1.0f / (1.0f + __expf(-0.125f * s[e]));
        int r = row0 + e + h16;
        __builtin_nontemporal_store(p, &attn[((size_t)(b * 3 + ph) * NSEQ + r) * NSEQ + col]);
        pw[(e + h16) * 40 + cg * 16 + (lane & 15)] = bf16_bits(p);
      }
    }
    union { v16bf v; uint4 q[2]; } pa;
    const unsigned short* pp = pw + (lane & 15) * 40 + h16;
    pa.q[0] = *(const uint4*)pp;
    pa.q[1] = *(const uint4*)(pp + 16);
#pragma unroll
    for (int dg = 0; dg < 4; ++dg)
      acc[dg] = wmma_bf16(pa.v, load_frag(vl, 32, dg * 16, 0), acc[dg]);
    __syncthreads();
    cur = nxt;
  }
#pragma unroll
  for (int dg = 0; dg < 4; ++dg) {
    int d = dg * 16 + (lane & 15);
    union { unsigned short u[8]; uint4 q; } pk;
#pragma unroll
    for (int e = 0; e < 8; ++e) pk.u[e] = bf16_bits(acc[dg][e]);
    *(uint4*)&c1t[((size_t)(b * 3 + ph) * HDIM + d) * NSEQ + row0 + h16] = pk.q;
  }
}

// ---- Pass B: ctx = softmax(Qp Kp^T/8) @ ctx1   (flash-style online softmax)
__global__ void k_attn_pose(const unsigned short* __restrict__ qb, const unsigned short* __restrict__ kb,
                            const unsigned short* __restrict__ c1t, unsigned short* __restrict__ ctxb) {
  __shared__ __align__(16) unsigned short ktile[2][32 * 64];
  __shared__ __align__(16) unsigned short vtile[2][64 * 32];
  __shared__ __align__(16) unsigned short pt[4 * 16 * 40];
  int tid = threadIdx.x;
  int warp = tid >> 5, lane = tid & 31;
  int rowt = blockIdx.x & 63;
  int hb_ = blockIdx.x >> 6;
  int ph = hb_ % 3, b = hb_ / 3;
  int row0 = rowt * 64 + warp * 16;
  const unsigned short* Q = qb + (size_t)(b * NH + ph + 3) * NSEQ * HDIM;
  const unsigned short* K = kb + (size_t)(b * NH + ph + 3) * NSEQ * HDIM;
  const unsigned short* V = c1t + (size_t)(b * 3 + ph) * HDIM * NSEQ;
  v16bf qa0 = load_frag(Q, HDIM, row0, 0);
  v16bf qa1 = load_frag(Q, HDIM, row0, 32);
  v8f z = {}; v8f acc[4] = {z, z, z, z};
  v8f mrun, lrun = z;
#pragma unroll
  for (int e = 0; e < 8; ++e) mrun[e] = -3.0e38f;
  int h16 = (lane >> 4) << 3;
  unsigned short* pw = &pt[warp * 640];
  int cur = 0;
  stage_tiles(&ktile[0][0], &vtile[0][0], K, V, 0, warp, tid);
  for (int m0 = 0; m0 < NSEQ; m0 += 32) {
    int nxt = cur ^ 1;
    if (m0 + 32 < NSEQ) {
      stage_tiles(&ktile[nxt][0], &vtile[nxt][0], K, V, m0 + 32, warp, tid);
#if HAVE_TDM
      __builtin_amdgcn_s_wait_tensorcnt((short)2);
#endif
    } else {
#if HAVE_TDM
      __builtin_amdgcn_s_wait_tensorcnt((short)0);
#endif
    }
    __syncthreads();
    const unsigned short* kt = &ktile[cur][0];
    const unsigned short* vl = &vtile[cur][0];
    v8f s0 = z, s1 = z;
    s0 = wmma_bf16(qa0, load_frag(kt, HDIM, 0, 0), s0);
    s0 = wmma_bf16(qa1, load_frag(kt, HDIM, 0, 32), s0);
    s1 = wmma_bf16(qa0, load_frag(kt, HDIM, 16, 0), s1);
    s1 = wmma_bf16(qa1, load_frag(kt, HDIM, 16, 32), s1);
    s0 = s0 * 0.125f; s1 = s1 * 0.125f;
    v8f alpha;
#pragma unroll
    for (int e = 0; e < 8; ++e) {
      float t = fmaxf(s0[e], s1[e]);
#pragma unroll
      for (int off = 8; off; off >>= 1) t = fmaxf(t, __shfl_xor(t, off, 16));
      float mn = fmaxf(mrun[e], t);
      float al = __expf(mrun[e] - mn);
      float p0 = __expf(s0[e] - mn), p1 = __expf(s1[e] - mn);
      float rs = p0 + p1;
#pragma unroll
      for (int off = 8; off; off >>= 1) rs += __shfl_xor(rs, off, 16);
      lrun[e] = lrun[e] * al + rs;
      mrun[e] = mn;
      alpha[e] = al;
      pw[(e + h16) * 40 + (lane & 15)] = bf16_bits(p0);
      pw[(e + h16) * 40 + 16 + (lane & 15)] = bf16_bits(p1);
    }
#pragma unroll
    for (int dg = 0; dg < 4; ++dg)
#pragma unroll
      for (int e = 0; e < 8; ++e) acc[dg][e] *= alpha[e];
    union { v16bf v; uint4 q[2]; } pa;
    const unsigned short* pp = pw + (lane & 15) * 40 + h16;
    pa.q[0] = *(const uint4*)pp;
    pa.q[1] = *(const uint4*)(pp + 16);
#pragma unroll
    for (int dg = 0; dg < 4; ++dg)
      acc[dg] = wmma_bf16(pa.v, load_frag(vl, 32, dg * 16, 0), acc[dg]);
    __syncthreads();
    cur = nxt;
  }
#pragma unroll
  for (int dg = 0; dg < 4; ++dg) {
    int d = dg * 16 + (lane & 15);
#pragma unroll
    for (int e = 0; e < 8; ++e) {
      int n = row0 + e + h16;
      float val = acc[dg][e] / lrun[e];
      ctxb[((size_t)b * NSEQ + n) * HALF + ph * HDIM + d] = bf16_bits(val);
    }
  }
}

// -------------------- proj GEMM + residual -> x1 (f32) --------------------
__global__ void k_proj(const unsigned short* __restrict__ ctxb, const unsigned short* __restrict__ wp,
                       const float* __restrict__ pb, const float* __restrict__ x,
                       float* __restrict__ x1) {
  int warp = threadIdx.x >> 5, lane = threadIdx.x & 31;
  int wid = blockIdx.x * 8 + warp;
  int rowt = wid / 6, cgrp = wid % 6;
  int row0 = rowt * 16, col0 = cgrp * 64;
  v8f z = {}; v8f acc[4] = {z, z, z, z};
  gemm16x64(ctxb, HALF, row0, wp, HALF, col0, 6, acc);
  int h16 = (lane >> 4) << 3;
#pragma unroll
  for (int f = 0; f < 4; ++f) {
    int o = col0 + f * 16 + (lane & 15);
#pragma unroll
    for (int e = 0; e < 8; ++e) {
      int t = row0 + e + h16;
      x1[(size_t)t * CDIM + o] = x[(size_t)t * CDIM + o] + acc[f][e] + pb[o];
    }
  }
}

// -------------------- fc1 GEMM + bias + exact GELU -> bf16 --------------------
__global__ void k_fc1(const unsigned short* __restrict__ h2b, const unsigned short* __restrict__ w1,
                      const float* __restrict__ b1, unsigned short* __restrict__ mo) {
  int warp = threadIdx.x >> 5, lane = threadIdx.x & 31;
  int wid = blockIdx.x * 8 + warp;
  int rowt = wid / 24, cgrp = wid % 24;
  int row0 = rowt * 16, col0 = cgrp * 64;
  v8f z = {}; v8f acc[4] = {z, z, z, z};
  gemm16x64(h2b, CDIM, row0, w1, CDIM, col0, 12, acc);
  int h16 = (lane >> 4) << 3;
#pragma unroll
  for (int f = 0; f < 4; ++f) {
    int o = col0 + f * 16 + (lane & 15);
#pragma unroll
    for (int e = 0; e < 8; ++e) {
      int t = row0 + e + h16;
      float xv = acc[f][e] + b1[o];
      float g = 0.5f * xv * (1.0f + erff(xv * 0.70710678f));
      mo[(size_t)t * MLPH + o] = bf16_bits(g);
    }
  }
}

// -------------------- fc2 GEMM + bias + residual -> out (f32) --------------------
__global__ void k_fc2(const unsigned short* __restrict__ mb, const unsigned short* __restrict__ w2,
                      const float* __restrict__ b2, const float* __restrict__ x1,
                      float* __restrict__ out) {
  int warp = threadIdx.x >> 5, lane = threadIdx.x & 31;
  int wid = blockIdx.x * 8 + warp;
  int rowt = wid / 6, cgrp = wid % 6;
  int row0 = rowt * 16, col0 = cgrp * 64;
  v8f z = {}; v8f acc[4] = {z, z, z, z};
  gemm16x64(mb, MLPH, row0, w2, MLPH, col0, 48, acc);
  int h16 = (lane >> 4) << 3;
#pragma unroll
  for (int f = 0; f < 4; ++f) {
    int o = col0 + f * 16 + (lane & 15);
#pragma unroll
    for (int e = 0; e < 8; ++e) {
      int t = row0 + e + h16;
      out[(size_t)t * CDIM + o] = x1[(size_t)t * CDIM + o] + acc[f][e] + b2[o];
    }
  }
}

extern "C" void kernel_launch(void* const* d_in, const int* in_sizes, int n_in,
                              void* d_out, int out_size, void* d_ws, size_t ws_size,
                              hipStream_t stream) {
  const float* x      = (const float*)d_in[0];
  const float* qkv_w  = (const float*)d_in[1];
  const float* proj_w = (const float*)d_in[2];
  const float* proj_b = (const float*)d_in[3];
  const float* g1     = (const float*)d_in[4];
  const float* b1     = (const float*)d_in[5];
  const float* g2     = (const float*)d_in[6];
  const float* b2     = (const float*)d_in[7];
  const float* fc1_w  = (const float*)d_in[8];
  const float* fc1_b  = (const float*)d_in[9];
  const float* fc2_w  = (const float*)d_in[10];
  const float* fc2_b  = (const float*)d_in[11];

  float* out0 = (float*)d_out;
  float* attn = out0 + (size_t)TOK * CDIM;   // second output: [B,3,N,N]

  char* ws = (char*)d_ws;
  size_t off = 0;
  auto alloc = [&](size_t bytes) { char* p = ws + off; off += bytes; return p; };
  unsigned short* hb   = (unsigned short*)alloc((size_t)TOK * CDIM * 2);
  unsigned short* h2b  = (unsigned short*)alloc((size_t)TOK * CDIM * 2);
  unsigned short* wqb  = (unsigned short*)alloc((size_t)3 * CDIM * CDIM * 2);
  unsigned short* w1b  = (unsigned short*)alloc((size_t)MLPH * CDIM * 2);
  unsigned short* w2b  = (unsigned short*)alloc((size_t)CDIM * MLPH * 2);
  unsigned short* wpb  = (unsigned short*)alloc((size_t)CDIM * HALF * 2);
  unsigned short* qbuf = (unsigned short*)alloc((size_t)2 * NH * NSEQ * HDIM * 2);
  unsigned short* kbuf = (unsigned short*)alloc((size_t)2 * NH * NSEQ * HDIM * 2);
  unsigned short* vt   = (unsigned short*)alloc((size_t)2 * 3 * HDIM * NSEQ * 2);
  unsigned short* c1t  = (unsigned short*)alloc((size_t)2 * 3 * HDIM * NSEQ * 2);
  unsigned short* ctxb = (unsigned short*)alloc((size_t)TOK * HALF * 2);
  float*          x1   = (float*)alloc((size_t)TOK * CDIM * 4);
  unsigned short* m1   = (unsigned short*)alloc((size_t)TOK * MLPH * 2);

  k_convert<<<256, 256, 0, stream>>>(qkv_w, wqb, 3 * CDIM * CDIM);
  k_convert<<<256, 256, 0, stream>>>(fc1_w, w1b, MLPH * CDIM);
  k_convert<<<256, 256, 0, stream>>>(fc2_w, w2b, CDIM * MLPH);
  k_convert<<<256, 256, 0, stream>>>(proj_w, wpb, CDIM * HALF);
  k_ln<<<TOK / 8, 256, 0, stream>>>(x, g1, b1, hb);
  k_qkv<<<1152, 256, 0, stream>>>(hb, wqb, qbuf, kbuf, vt);
  k_attn_obj<<<384, 128, 0, stream>>>(qbuf, kbuf, vt, attn, c1t);
  k_attn_pose<<<384, 128, 0, stream>>>(qbuf, kbuf, c1t, ctxb);
  k_proj<<<384, 256, 0, stream>>>(ctxb, wpb, proj_b, x, x1);
  k_ln<<<TOK / 8, 256, 0, stream>>>(x1, g2, b2, h2b);
  k_fc1<<<1536, 256, 0, stream>>>(h2b, w1b, fc1_b, m1);
  k_fc2<<<384, 256, 0, stream>>>(m1, w2b, fc2_b, x1, out0);
}
